// SparseDeformableMambaBlock_85676007620946
// MI455X (gfx1250) — matrix operand order repaired and checked by reading, not compiled, not run on hardware
//
#include <hip/hip_runtime.h>

// ---------------- Problem constants (from reference) ----------------
#define DIMC   512
#define DSTATE 16
#define DCONV  4
#define EXPD   1024        // EXP_DIM
#define BATCH  4
#define SEQ    4096
#define KSEL   1228        // max(1, int(4096*0.3))
#define EPS_RMS  1e-6f
#define EPS_NORM 1e-12f

// ---------------- WMMA types ----------------
typedef __attribute__((ext_vector_type(16))) __bf16 v16bf;
typedef __attribute__((ext_vector_type(8)))  float  v8f;

union FragAB { v16bf v; uint4 q[2]; unsigned u[8]; };

__device__ __forceinline__ unsigned short f2bf(float f) {
    unsigned u = __float_as_uint(f);
    unsigned r = (u + 0x7FFFu + ((u >> 16) & 1u)) >> 16;   // RNE
    return (unsigned short)r;
}

// A fragment: 16x32 bf16, rows m0..m0+15, cols k0..k0+31, row-major src [.. x lda].
// Per ISA 7.12.2, lane (m,grp) needs shorts k = 8*grp + {0..7} and 16 + 8*grp + {0..7}:
// two contiguous 16B chunks -> two global_load_b128, no divergence. OOB rows are
// clamped (their D rows are discarded at store time), never zero-filled.
__device__ __forceinline__ void load_a_frag(FragAB& f, const unsigned short* __restrict__ A,
                                            int m0, int k0, int lda, int lane, int Mv) {
    int m   = m0 + (lane & 15);
    m = (m < Mv) ? m : (Mv - 1);               // clamp: no branch, no zero-fill needed
    int grp = lane >> 4;
    const unsigned short* p = A + (size_t)m * lda + k0 + 8 * grp;
    f.q[0] = *(const uint4*)(p);
    f.q[1] = *(const uint4*)(p + 16);
}

// B fragment: 32x16 bf16 (KxN). Source is weight [N x K] row-major (K contiguous).
// lane (n,grp) holds shorts k = 8*grp + {0..7} and 16 + 8*grp + {0..7}: two 16B chunks.
__device__ __forceinline__ void load_b_frag(FragAB& f, const unsigned short* __restrict__ Bw,
                                            int n0, int k0, int ldb, int lane) {
    int n   = n0 + (lane & 15);
    int grp = lane >> 4;
    const unsigned short* p = Bw + (size_t)n * ldb + k0 + 8 * grp;
    f.q[0] = *(const uint4*)(p);
    f.q[1] = *(const uint4*)(p + 16);
}

// C/D fragment store: VGPR r -> M = m0 + r + 8*grp, N = n0 + (lane&15)
__device__ __forceinline__ void store_c_frag(float* __restrict__ C, const v8f& c,
                                             int m0, int n0, int ldc, int lane, int Mv) {
    int n  = n0 + (lane & 15);
    int mb = m0 + 8 * (lane >> 4);
    if (m0 + 31 < Mv) {                        // wave-uniform fast path (no per-row cmp)
#pragma unroll
        for (int r = 0; r < 8; ++r) C[(size_t)(mb + r) * ldc + n] = c[r];
    } else {
#pragma unroll
        for (int r = 0; r < 8; ++r) {
            int m = mb + r;
            if (m < Mv) C[(size_t)m * ldc + n] = c[r];
        }
    }
}

// ---------------- Kernels ----------------

__global__ void k_f32_to_bf16(const float* __restrict__ src, unsigned short* __restrict__ dst, int n) {
    int i = blockIdx.x * blockDim.x + threadIdx.x;
    if (i < n) dst[i] = f2bf(src[i]);
}

// RMSNorm row -> bf16 activations. One block (256 thr) per row of 512.
__global__ void k_rmsnorm(const float* __restrict__ x, const float* __restrict__ nw,
                          unsigned short* __restrict__ xn) {
    int row = blockIdx.x;
    int tid = threadIdx.x;
    const float* xr = x + (size_t)row * DIMC;
    float v0 = xr[tid], v1 = xr[tid + 256];
    __shared__ float red[256];
    red[tid] = v0 * v0 + v1 * v1;
    __syncthreads();
    for (int off = 128; off > 0; off >>= 1) { if (tid < off) red[tid] += red[tid + off]; __syncthreads(); }
    float scale = rsqrtf(red[0] * (1.0f / DIMC) + EPS_RMS);
    unsigned short* o = xn + (size_t)row * DIMC;
    o[tid]       = f2bf(v0 * scale * nw[tid]);
    o[tid + 256] = f2bf(v1 * scale * nw[tid + 256]);
}

// Generic bf16 WMMA GEMM: C[MxN] = A[MxK] * Bw[NxK]^T + bias.
// One wave per 32x32 tile; 4 x v_wmma_f32_16x16x32_bf16 per K-step of 32.
__global__ void k_gemm(const unsigned short* __restrict__ A,
                       const unsigned short* __restrict__ Bw,
                       const float* __restrict__ bias,
                       float* __restrict__ C,
                       int Mv, int N, int K, int tiles_n, int total_tiles) {
    int wave = (blockIdx.x * blockDim.x + threadIdx.x) >> 5;
    int lane = threadIdx.x & 31;
    if (wave >= total_tiles) return;            // wave-uniform
    int tm = wave / tiles_n, tn = wave % tiles_n;
    int m0 = tm * 32, n0 = tn * 32;

    int nA = n0 + (lane & 15);
    float bz0 = bias[nA];
    float bz1 = bias[nA + 16];
    v8f c00, c01, c10, c11;
#pragma unroll
    for (int r = 0; r < 8; ++r) { c00[r] = bz0; c01[r] = bz1; c10[r] = bz0; c11[r] = bz1; }

    for (int k0 = 0; k0 < K; k0 += 32) {
        FragAB a0, a1, b0, b1;
        load_a_frag(a0, A, m0,      k0, K, lane, Mv);
        load_a_frag(a1, A, m0 + 16, k0, K, lane, Mv);
        load_b_frag(b0, Bw, n0,      k0, K, lane);
        load_b_frag(b1, Bw, n0 + 16, k0, K, lane);
        c00 = __builtin_amdgcn_wmma_f32_16x16x32_bf16(false, a0.v, false, b0.v, (short)0, c00, false, false);
        c01 = __builtin_amdgcn_wmma_f32_16x16x32_bf16(false, a0.v, false, b1.v, (short)0, c01, false, false);
        c10 = __builtin_amdgcn_wmma_f32_16x16x32_bf16(false, a1.v, false, b0.v, (short)0, c10, false, false);
        c11 = __builtin_amdgcn_wmma_f32_16x16x32_bf16(false, a1.v, false, b1.v, (short)0, c11, false, false);
    }
    store_c_frag(C, c00, m0,      n0,      N, lane, Mv);
    store_c_frag(C, c01, m0,      n0 + 16, N, lane, Mv);
    store_c_frag(C, c10, m0 + 16, n0,      N, lane, Mv);
    store_c_frag(C, c11, m0 + 16, n0 + 16, N, lane, Mv);
}

// Cosine similarity of each xp row against the center row (softmax skipped: monotone).
__global__ void k_sim(const float* __restrict__ xp, float* __restrict__ sim) {
    int row = blockIdx.x;
    int b = row / SEQ;
    int tid = threadIdx.x;
    const float* xr = xp + (size_t)row * EXPD;
    const float* cr = xp + ((size_t)b * SEQ + SEQ / 2) * EXPD;
    float dot = 0.f, ss = 0.f, cs = 0.f;
    for (int i = tid; i < EXPD; i += 256) {
        float a = xr[i], c = cr[i];
        dot += a * c; ss += a * a; cs += c * c;
    }
    __shared__ float rd[256], rs[256], rc[256];
    rd[tid] = dot; rs[tid] = ss; rc[tid] = cs;
    __syncthreads();
    for (int off = 128; off > 0; off >>= 1) {
        if (tid < off) { rd[tid] += rd[tid + off]; rs[tid] += rs[tid + off]; rc[tid] += rc[tid + off]; }
        __syncthreads();
    }
    if (tid == 0) {
        float nr = fmaxf(sqrtf(rs[0]), EPS_NORM);
        float nc = fmaxf(sqrtf(rc[0]), EPS_NORM);
        sim[row] = rd[0] / (nr * nc);
    }
}

// Exact stable top-k via descending rank (ties: lower index first) — matches lax.top_k order.
__global__ void k_topk(const float* __restrict__ sim, int* __restrict__ idx) {
    int b = blockIdx.x;
    __shared__ float s[SEQ];
    int tid = threadIdx.x;                      // 1024 threads
    for (int i = tid; i < SEQ; i += 1024) s[i] = sim[b * SEQ + i];
    __syncthreads();
    for (int l = tid; l < SEQ; l += 1024) {
        float v = s[l];
        int rank = 0;
        for (int j = 0; j < SEQ; ++j) {
            float sj = s[j];                    // LDS broadcast (all lanes same addr)
            rank += (sj > v) || (sj == v && j < l);
        }
        if (rank < KSEL) idx[b * KSEL + rank] = l;
    }
}

// Gathered causal depthwise conv: xc[b,j,d] = sum_t cw[d,t] * xp[b, idx[b,j-3+t], d]
__global__ void k_conv(const float* __restrict__ xp, const int* __restrict__ idx,
                       const float* __restrict__ cw, float* __restrict__ xc) {
    int bj = blockIdx.x;
    int b = bj / KSEL, j = bj % KSEL;
    __shared__ int sidx[DCONV];
    if (threadIdx.x < DCONV) {
        int jj = j - (DCONV - 1) + (int)threadIdx.x;
        sidx[threadIdx.x] = (jj >= 0) ? idx[b * KSEL + jj] : -1;
    }
    __syncthreads();
    for (int d = threadIdx.x; d < EXPD; d += 256) {
        float acc = 0.f;
#pragma unroll
        for (int t = 0; t < DCONV; ++t) {
            int l = sidx[t];
            if (l >= 0) acc += cw[d * DCONV + t] * xp[((size_t)b * SEQ + l) * EXPD + d];
        }
        xc[((size_t)b * KSEL + j) * EXPD + d] = acc;
    }
}

// SSM scan: 16 lanes per (b,d) chain, state index i per lane, A-row in registers,
// cross-lane h gathered with width-16 shuffles. Output written as bf16 for GEMM2.
__global__ void k_scan(const float* __restrict__ xc, const float* __restrict__ A,
                       const float* __restrict__ Bp, const float* __restrict__ Cp,
                       unsigned short* __restrict__ outs) {
    int tid = threadIdx.x;
    int chain = blockIdx.x * (blockDim.x >> 4) + (tid >> 4);
    int i = tid & 15;
    int b = chain >> 10;            // / EXPD
    int d = chain & 1023;
    float a[16];
#pragma unroll
    for (int j = 0; j < 16; ++j) a[j] = A[i * DSTATE + j];
    float sB = 1.f / (1.f + __expf(-Bp[i]));
    float sC = 1.f / (1.f + __expf(-Cp[d * DSTATE + i]));
    float h = 0.f;
    const float* xrow = xc + (size_t)b * KSEL * EXPD + d;
    unsigned short* orow = outs + (size_t)b * KSEL * EXPD + d;
    for (int t = 0; t < KSEL; ++t) {
        float xt = xrow[(size_t)t * EXPD];
        float hn = sB * xt;
#pragma unroll
        for (int j = 0; j < 16; ++j)
            hn = fmaf(__shfl(h, j, 16), a[j], hn);   // h_i = sum_j h_j * A[i][j] + sB_i*x
        h = hn;
        float y = hn * sC;
#pragma unroll
        for (int off = 8; off > 0; off >>= 1) y += __shfl_down(y, off, 16);
        if (i == 0) orow[(size_t)t * EXPD] = f2bf(y);
    }
}

__global__ void k_copy4(const float4* __restrict__ src, float4* __restrict__ dst, int n) {
    int i = blockIdx.x * blockDim.x + threadIdx.x;
    if (i < n) dst[i] = src[i];
}

// out[b, idx[b,j], :] += xproc[b, j, :]   (out already holds x; indices are unique)
__global__ void k_scatter(const float* __restrict__ xproc, const int* __restrict__ idx,
                          float* __restrict__ out) {
    int bj = blockIdx.x;
    int b = bj / KSEL;
    int l = idx[bj];
    const float* src = xproc + (size_t)bj * DIMC;
    float* dst = out + ((size_t)b * SEQ + l) * DIMC;
    for (int c = threadIdx.x; c < DIMC; c += 256) dst[c] += src[c];
}

// ---------------- Host launcher ----------------
extern "C" void kernel_launch(void* const* d_in, const int* in_sizes, int n_in,
                              void* d_out, int out_size, void* d_ws, size_t ws_size,
                              hipStream_t stream) {
    const float* x      = (const float*)d_in[0];
    const float* norm_w = (const float*)d_in[1];
    const float* w_in   = (const float*)d_in[2];
    const float* b_in   = (const float*)d_in[3];
    const float* w_out  = (const float*)d_in[4];
    const float* b_out  = (const float*)d_in[5];
    const float* Amat   = (const float*)d_in[6];
    const float* Bp     = (const float*)d_in[7];
    const float* Cp     = (const float*)d_in[8];
    const float* conv_w = (const float*)d_in[9];
    float* out = (float*)d_out;

    // Workspace carve-out (aligned 256B). Total ~127 MB.
    char* w = (char*)d_ws;
    auto alloc = [&](size_t bytes) -> void* {
        void* p = (void*)w;
        w += (bytes + 255) & ~(size_t)255;
        return p;
    };
    unsigned short* ws_xn    = (unsigned short*)alloc((size_t)BATCH * SEQ * DIMC * 2);
    unsigned short* ws_winb  = (unsigned short*)alloc((size_t)EXPD * DIMC * 2);
    unsigned short* ws_woutb = (unsigned short*)alloc((size_t)DIMC * EXPD * 2);
    float*          ws_xp    = (float*)alloc((size_t)BATCH * SEQ * EXPD * 4);
    float*          ws_sim   = (float*)alloc((size_t)BATCH * SEQ * 4);
    int*            ws_idx   = (int*)alloc((size_t)BATCH * KSEL * 4);
    float*          ws_xc    = (float*)alloc((size_t)BATCH * KSEL * EXPD * 4);
    unsigned short* ws_outs  = (unsigned short*)alloc((size_t)BATCH * KSEL * EXPD * 2);
    float*          ws_xproc = (float*)alloc((size_t)BATCH * KSEL * DIMC * 4);

    // 1) weights -> bf16
    k_f32_to_bf16<<<(EXPD * DIMC + 255) / 256, 256, 0, stream>>>(w_in,  ws_winb,  EXPD * DIMC);
    k_f32_to_bf16<<<(DIMC * EXPD + 255) / 256, 256, 0, stream>>>(w_out, ws_woutb, DIMC * EXPD);

    // 2) RMSNorm -> bf16 activations
    k_rmsnorm<<<BATCH * SEQ, 256, 0, stream>>>(x, norm_w, ws_xn);

    // 3) GEMM1: xp = xn @ w_in^T + b_in   (M=16384,N=1024,K=512; 32x32 wave tiles)
    {
        int tiles_m = (BATCH * SEQ) / 32, tiles_n = EXPD / 32;
        int total = tiles_m * tiles_n;                 // 16384 waves
        k_gemm<<<(total + 7) / 8, 256, 0, stream>>>(ws_xn, ws_winb, b_in, ws_xp,
                                                    BATCH * SEQ, EXPD, DIMC, tiles_n, total);
    }

    // 4) cosine similarity vs center row
    k_sim<<<BATCH * SEQ, 256, 0, stream>>>(ws_xp, ws_sim);

    // 5) top-k indices (descending, stable)
    k_topk<<<BATCH, 1024, 0, stream>>>(ws_sim, ws_idx);

    // 6) gathered depthwise causal conv
    k_conv<<<BATCH * KSEL, 256, 0, stream>>>(ws_xp, ws_idx, conv_w, ws_xc);

    // 7) SSM scan (16 lanes / chain, 4096 chains)
    k_scan<<<(BATCH * EXPD * 16) / 256, 256, 0, stream>>>(ws_xc, Amat, Bp, Cp, ws_outs);

    // 8) GEMM2 per batch: xproc = outs @ w_out^T + b_out  (M=1228,N=512,K=1024)
    {
        int tiles_m = (KSEL + 31) / 32, tiles_n = DIMC / 32;
        int total = tiles_m * tiles_n;                 // 624 waves per batch
        for (int b = 0; b < BATCH; ++b) {
            k_gemm<<<(total + 7) / 8, 256, 0, stream>>>(
                ws_outs + (size_t)b * KSEL * EXPD, ws_woutb, b_out,
                ws_xproc + (size_t)b * KSEL * DIMC, KSEL, DIMC, EXPD, tiles_n, total);
        }
    }

    // 9) residual copy, 10) scatter-add of processed rows
    {
        int n4 = (BATCH * SEQ * DIMC) / 4;
        k_copy4<<<(n4 + 255) / 256, 256, 0, stream>>>((const float4*)x, (float4*)out, n4);
    }
    k_scatter<<<BATCH * KSEL, 256, 0, stream>>>(ws_xproc, ws_idx, out);
}
